// VarianceAdaptor_38603166056644
// MI455X (gfx1250) — matrix operand order repaired and checked by reading, not compiled, not run on hardware
//
#include <hip/hip_runtime.h>
#include <hip/hip_bf16.h>

// Problem constants (from reference)
#define BB  16
#define SS  512
#define HH  256
#define FSN 256
#define TT  4096
#define NBN 256
#define NSN 10

typedef __attribute__((ext_vector_type(16))) __bf16 v16bf;
typedef __attribute__((ext_vector_type(8)))  float  v8f;

union Frag { v16bf v; float4 q[2]; };

#if __has_builtin(__builtin_amdgcn_global_load_async_to_lds_b32)
#define USE_ASYNC_LDS 1
typedef __attribute__((address_space(1))) int* gint_ptr;
typedef __attribute__((address_space(3))) int* lint_ptr;
#else
#define USE_ASYNC_LDS 0
#endif

__device__ __forceinline__ unsigned short f2bf(float f) {
  unsigned int u = __float_as_uint(f);
  u += 0x7FFFu + ((u >> 16) & 1u);       // round-to-nearest-even
  return (unsigned short)(u >> 16);
}
__device__ __forceinline__ float bf2f(unsigned short h) {
  return __uint_as_float(((unsigned int)h) << 16);
}

__device__ __forceinline__ void wait_async0() {
#if USE_ASYNC_LDS
#if __has_builtin(__builtin_amdgcn_s_wait_asynccnt)
  __builtin_amdgcn_s_wait_asynccnt(0);
#else
  asm volatile("s_wait_asynccnt 0x0" ::: "memory");
#endif
#endif
}

// ---------------------------------------------------------------------------
// f32 -> bf16 cast
__global__ __launch_bounds__(256) void k_cast(const float* __restrict__ in,
                                              unsigned short* __restrict__ out, int n) {
  int i = blockIdx.x * 256 + threadIdx.x;
  if (i < n) out[i] = f2bf(in[i]);
}

// ---------------------------------------------------------------------------
// Pack conv weights (Cout=256, Cin=256, K=3) into WMMA B-fragment order:
// wp[ntile(16)][kb(24)][512], fragment element p = lane*16 + j:
//   lane<16 : N=lane,    Kd=j        lane>=16 : N=lane-16, Kd=16+j
// kb = kk*8 + cb ; channel c = cb*32 + Kd ; tap kk ; f = ntile*16 + N
__global__ __launch_bounds__(256) void k_pack(const float* __restrict__ w,
                                              unsigned short* __restrict__ wp) {
  int idx = blockIdx.x * 256 + threadIdx.x;
  const int total = 16 * 24 * 512;
  if (idx >= total) return;
  int p    = idx & 511;
  int kb   = (idx >> 9) % 24;
  int nt   = idx / (512 * 24);
  int lane = p >> 4;
  int j    = p & 15;
  int kd   = (lane < 16) ? j : (16 + j);
  int n    = (lane < 16) ? lane : (lane - 16);
  int kk   = kb >> 3;
  int cb   = kb & 7;
  int c    = cb * 32 + kd;
  int f    = nt * 16 + n;
  wp[idx] = f2bf(w[((size_t)f * 256 + c) * 3 + kk]);
}

// ---------------------------------------------------------------------------
// Fused conv1d(K=3,pad=1) + bias + relu + LayerNorm, Cin=Cout=256, bf16 I/O.
// One block = 16 sequence rows x 256 output channels. 8 waves, 2 N-tiles/wave.
// Per channel block, an 18x32 activation slab (rows s0-1..s0+16) is loaded to
// LDS once (async-to-LDS when available) and reused by all 3 conv taps.
__global__ __launch_bounds__(256) void k_conv_ln(
    const unsigned short* __restrict__ act,   // bf16 (rows, 256)
    const unsigned short* __restrict__ wpack, // packed bf16 weights
    const float* __restrict__ bias,
    const float* __restrict__ gamma,
    const float* __restrict__ beta,
    unsigned short* __restrict__ out,         // bf16 (rows, 256)
    int seq)                                  // rows per batch (512 or 4096)
{
  __shared__ __align__(16) unsigned short sA[18 * 32];
  __shared__ float sC[16 * 256];
  __shared__ float sPart[16 * 16 * 2];
  __shared__ float sMean[16];
  __shared__ float sRstd[16];

  const int tid  = threadIdx.x;
  const int lane = tid & 31;
  const int wave = tid >> 5;
  const int m0   = blockIdx.x * 16;
  const int b    = m0 / seq;
  const int s0   = m0 % seq;

  const int nt0 = wave * 2, nt1 = wave * 2 + 1;
  v8f c0 = {}; v8f c1 = {};

  for (int cb = 0; cb < 8; ++cb) {
    // ---- fill 18x32 bf16 slab: 288 dwords, threads cover d = tid, 256+tid
    const unsigned short* actcb = act + (size_t)b * seq * 256 + cb * 32;
#pragma unroll
    for (int rep = 0; rep < 2; ++rep) {
      int d = rep * 256 + tid;
      if (d < 18 * 16) {
        int row  = d >> 4;        // 0..17  -> s = s0 + row - 1
        int col2 = d & 15;        // dword within row
        int s    = s0 + row - 1;
        if (s >= 0 && s < seq) {
#if USE_ASYNC_LDS
          __builtin_amdgcn_global_load_async_to_lds_b32(
              (gint_ptr)(actcb + (size_t)s * 256 + col2 * 2),
              (lint_ptr)&sA[d * 2], 0, 0);
#else
          *(unsigned int*)&sA[d * 2] =
              *(const unsigned int*)(actcb + (size_t)s * 256 + col2 * 2);
#endif
        } else {
          *(unsigned int*)&sA[d * 2] = 0u;   // conv zero-padding rows
        }
      }
    }
    wait_async0();
    __syncthreads();

    // ---- 3 taps x 2 N-tiles of WMMA from the shared slab
#pragma unroll
    for (int kk = 0; kk < 3; ++kk) {
      Frag a;
      {
        int m   = lane & 15;
        int off = (lane < 16) ? 0 : 16;
        a.q[0] = *(const float4*)&sA[(m + kk) * 32 + off];
        a.q[1] = *(const float4*)&sA[(m + kk) * 32 + off + 8];
      }
      const int kb = kk * 8 + cb;
      Frag b0, b1;
      {
        const float4* w0 = (const float4*)(wpack + ((size_t)(nt0 * 24 + kb) * 512 + lane * 16));
        const float4* w1 = (const float4*)(wpack + ((size_t)(nt1 * 24 + kb) * 512 + lane * 16));
        b0.q[0] = w0[0]; b0.q[1] = w0[1];
        b1.q[0] = w1[0]; b1.q[1] = w1[1];
      }
      if (kk == 0 && cb + 1 < 8)
        __builtin_prefetch(wpack + ((size_t)(nt0 * 24 + cb + 1) * 512 + lane * 16), 0, 0);
      c0 = __builtin_amdgcn_wmma_f32_16x16x32_bf16(false, a.v, false, b0.v, (short)0, c0, false, false);
      c1 = __builtin_amdgcn_wmma_f32_16x16x32_bf16(false, a.v, false, b1.v, (short)0, c1, false, false);
    }
    __syncthreads();
  }

  // bias + relu -> LDS (C layout: vgpr r -> M = r + 8*(lane>>4), N = lane&15)
  {
    const int col0 = nt0 * 16 + (lane & 15);
    const int col1 = nt1 * 16 + (lane & 15);
    const float bs0 = bias[col0], bs1 = bias[col1];
    const int rbase = (lane >> 4) * 8;
    for (int r = 0; r < 8; ++r) {
      float v0 = c0[r] + bs0; v0 = v0 > 0.f ? v0 : 0.f;
      float v1 = c1[r] + bs1; v1 = v1 > 0.f ? v1 : 0.f;
      sC[(rbase + r) * 256 + col0] = v0;
      sC[(rbase + r) * 256 + col1] = v1;
    }
  }
  __syncthreads();

  // LayerNorm over 256 channels per row
  {
    int row = tid >> 4, g = tid & 15;
    float s = 0.f, ss = 0.f;
    for (int i = 0; i < 16; ++i) {
      float v = sC[row * 256 + g * 16 + i];
      s += v; ss += v * v;
    }
    sPart[(row * 16 + g) * 2 + 0] = s;
    sPart[(row * 16 + g) * 2 + 1] = ss;
  }
  __syncthreads();
  if (tid < 16) {
    float s = 0.f, ss = 0.f;
    for (int g = 0; g < 16; ++g) {
      s  += sPart[(tid * 16 + g) * 2 + 0];
      ss += sPart[(tid * 16 + g) * 2 + 1];
    }
    float m   = s * (1.f / 256.f);
    float var = ss * (1.f / 256.f) - m * m;
    sMean[tid] = m;
    sRstd[tid] = rsqrtf(var + 1e-5f);
  }
  __syncthreads();
  {
    int row = tid >> 4, g = tid & 15;
    float m = sMean[row], rs = sRstd[row];
    size_t obase = (size_t)(m0 + row) * 256;
    for (int i = 0; i < 16; ++i) {
      int c = g * 16 + i;
      float v = (sC[row * 256 + c] - m) * rs * gamma[c] + beta[c];
      out[obase + c] = f2bf(v);
    }
  }
}

// ---------------------------------------------------------------------------
// Predictor head: out[row] = mask ? 0 : dot(h[row,:256], lw) + lb
__global__ __launch_bounds__(256) void k_rowdot(const unsigned short* __restrict__ h,
                                                const float* __restrict__ lw,
                                                const float* __restrict__ lb,
                                                const unsigned char* __restrict__ mask,
                                                float* __restrict__ out, int nrows) {
  int row  = blockIdx.x * 8 + (threadIdx.x >> 5);
  int lane = threadIdx.x & 31;
  if (row >= nrows) return;
  float s = 0.f;
  for (int i = 0; i < 8; ++i) {
    int c = i * 32 + lane;
    s += bf2f(h[(size_t)row * 256 + c]) * lw[c];
  }
  for (int o = 16; o; o >>= 1) s += __shfl_down(s, o, 32);
  if (lane == 0) out[row] = mask[row] ? 0.f : (s + lb[0]);
}

// ---------------------------------------------------------------------------
__global__ __launch_bounds__(256) void k_eidx(const float* __restrict__ et,
                                              const float* __restrict__ bins,
                                              int* __restrict__ eidx, int n) {
  int i = blockIdx.x * 256 + threadIdx.x;
  if (i >= n) return;
  float v = et[i];
  int lo = 0, hi = NBN - 1;                 // 255 bins, 'left' insertion
  while (lo < hi) { int mid = (lo + hi) >> 1; if (bins[mid] < v) lo = mid + 1; else hi = mid; }
  eidx[i] = lo;
}

__global__ void k_cum(const int* __restrict__ dur, int* __restrict__ cum,
                      int* __restrict__ mlen_i, float* __restrict__ out_mlen) {
  int b = threadIdx.x;
  if (b >= BB) return;
  int s = 0;
  for (int i = 0; i < SS; ++i) { s += dur[b * SS + i]; cum[b * SS + i] = s; }
  mlen_i[b] = s;
  out_mlen[b] = (float)s;
}

__global__ __launch_bounds__(256) void k_addemb(const float* __restrict__ x,
                                                const float* __restrict__ emb,
                                                const int* __restrict__ eidx,
                                                float* __restrict__ xq, int n) {
  int i = blockIdx.x * 256 + threadIdx.x;
  if (i >= n) return;
  int row = i >> 8, c = i & 255;
  xq[i] = x[i] + emb[(size_t)eidx[row] * HH + c];
}

__global__ __launch_bounds__(256) void k_expand_idx(const int* __restrict__ cum,
                                                    const int* __restrict__ mlen,
                                                    int* __restrict__ fidx,
                                                    int* __restrict__ vmask, int n) {
  int i = blockIdx.x * 256 + threadIdx.x;
  if (i >= n) return;
  int b = i / TT, t = i % TT;
  const int* c = cum + b * SS;
  int lo = 0, hi = SS;                      // side='right'
  while (lo < hi) { int mid = (lo + hi) >> 1; if (c[mid] <= t) lo = mid + 1; else hi = mid; }
  fidx[i]  = lo > SS - 1 ? SS - 1 : lo;
  vmask[i] = (t < mlen[b]) ? 1 : 0;
}

__global__ __launch_bounds__(256) void k_gather(const float* __restrict__ xq,
                                                const int* __restrict__ fidx,
                                                const int* __restrict__ vmask,
                                                unsigned short* __restrict__ xexp, int n) {
  int i = blockIdx.x * 256 + threadIdx.x;
  if (i >= n) return;
  int row = i >> 8, c = i & 255, b = row / TT;
  float v = vmask[row] ? xq[((size_t)(b * SS + fidx[row])) * HH + c] : 0.f;
  xexp[i] = f2bf(v);
}

// ---------------------------------------------------------------------------
// pitch_pred = h2 @ pp_w.T + pp_b ; pitch1d = sum_j pitch_pred*icwt
__global__ __launch_bounds__(256) void k_pitch(const unsigned short* __restrict__ h2,
                                               const float* __restrict__ pp_w,
                                               const float* __restrict__ pp_b,
                                               const float* __restrict__ icwt,
                                               float* __restrict__ pitch_out,
                                               float* __restrict__ pitch1d, int nrows) {
  __shared__ float sW[NSN * 256];
  for (int i = threadIdx.x; i < NSN * 256; i += 256) sW[i] = pp_w[i];
  __syncthreads();
  int row  = blockIdx.x * 8 + (threadIdx.x >> 5);
  int lane = threadIdx.x & 31;
  if (row >= nrows) return;
  float acc[NSN];
  for (int j = 0; j < NSN; ++j) acc[j] = 0.f;
  for (int i = 0; i < 8; ++i) {
    int c = i * 32 + lane;
    float hv = bf2f(h2[(size_t)row * 256 + c]);
    for (int j = 0; j < NSN; ++j) acc[j] += hv * sW[j * 256 + c];
  }
  for (int j = 0; j < NSN; ++j)
    for (int o = 16; o; o >>= 1) acc[j] += __shfl_down(acc[j], o, 32);
  if (lane == 0) {
    float pd = 0.f;
    for (int j = 0; j < NSN; ++j) {
      float p = acc[j] + pp_b[j];
      pitch_out[(size_t)row * NSN + j] = p;
      pd += p * icwt[j];
    }
    pitch1d[row] = pd;
  }
}

__global__ void k_hsum(const unsigned short* __restrict__ h2, float* __restrict__ sbh) {
  int b = blockIdx.x, c = threadIdx.x;
  float s = 0.f;
  const unsigned short* p = h2 + (size_t)b * TT * 256 + c;
  for (int t = 0; t < TT; ++t) s += bf2f(p[(size_t)t * 256]);
  sbh[b * 256 + c] = s;
}

__global__ void k_meanvar(const float* __restrict__ sbh, const float* __restrict__ ps_w,
                          const float* __restrict__ ps_b, float* __restrict__ out) {
  int t = threadIdx.x;
  if (t >= BB * 2) return;
  int b = t >> 1, j = t & 1;
  float s = 0.f;
  for (int c = 0; c < 256; ++c) s += sbh[b * 256 + c] * ps_w[j * 256 + c];
  out[b * 2 + j] = s * (1.0f / TT) + ps_b[j];
}

__global__ __launch_bounds__(256) void k_xout(const float* __restrict__ xq,
                                              const int* __restrict__ fidx,
                                              const int* __restrict__ vmask,
                                              const float* __restrict__ p1d,
                                              float* __restrict__ out, int n) {
  int i = blockIdx.x * 256 + threadIdx.x;
  if (i >= n) return;
  int row = i >> 8, c = i & 255, b = row / TT;
  float v = vmask[row] ? xq[((size_t)(b * SS + fidx[row])) * HH + c] : 0.f;
  out[i] = v + p1d[row];
}

__global__ __launch_bounds__(256) void k_misc(const int* __restrict__ dur,
                                              const unsigned char* __restrict__ mmask,
                                              float* __restrict__ out_dur,
                                              float* __restrict__ out_mask) {
  int i = blockIdx.x * 256 + threadIdx.x;
  if (i < BB * SS) out_dur[i] = (float)dur[i];
  if (i < BB * TT) out_mask[i] = mmask[i] ? 1.f : 0.f;
}

// ---------------------------------------------------------------------------
extern "C" void kernel_launch(void* const* d_in, const int* in_sizes, int n_in,
                              void* d_out, int out_size, void* d_ws, size_t ws_size,
                              hipStream_t stream) {
  (void)in_sizes; (void)n_in; (void)out_size; (void)ws_size;
  const float* x        = (const float*)d_in[0];
  const float* etgt     = (const float*)d_in[1];
  const int*   dur      = (const int*)d_in[2];
  const unsigned char* src_mask = (const unsigned char*)d_in[3];
  const unsigned char* mel_mask = (const unsigned char*)d_in[4];
  const float* bins     = (const float*)d_in[6];
  const float* emb      = (const float*)d_in[7];
  const float* conv_w[6] = { (const float*)d_in[8],  (const float*)d_in[12],
                             (const float*)d_in[18], (const float*)d_in[22],
                             (const float*)d_in[28], (const float*)d_in[32] };
  const float* conv_b[6] = { (const float*)d_in[9],  (const float*)d_in[13],
                             (const float*)d_in[19], (const float*)d_in[23],
                             (const float*)d_in[29], (const float*)d_in[33] };
  const float* conv_g[6] = { (const float*)d_in[10], (const float*)d_in[14],
                             (const float*)d_in[20], (const float*)d_in[24],
                             (const float*)d_in[30], (const float*)d_in[34] };
  const float* conv_be[6]= { (const float*)d_in[11], (const float*)d_in[15],
                             (const float*)d_in[21], (const float*)d_in[25],
                             (const float*)d_in[31], (const float*)d_in[35] };
  const float* dp_lw = (const float*)d_in[16];
  const float* dp_lb = (const float*)d_in[17];
  const float* ep_lw = (const float*)d_in[26];
  const float* ep_lb = (const float*)d_in[27];
  const float* pp_w  = (const float*)d_in[36];
  const float* pp_b  = (const float*)d_in[37];
  const float* icwt  = (const float*)d_in[38];
  const float* ps_w  = (const float*)d_in[39];
  const float* ps_b  = (const float*)d_in[40];

  const size_t BS  = (size_t)BB * SS;
  const size_t BT  = (size_t)BB * TT;
  const size_t BSH = BS * HH;
  const size_t BTH = BT * HH;

  // workspace bump allocator
  char* base = (char*)d_ws;
  size_t off = 0;
  auto alloc = [&](size_t bytes) -> void* {
    off = (off + 255) & ~(size_t)255;
    void* p = base + off;
    off += bytes;
    return p;
  };
  unsigned short* xbf  = (unsigned short*)alloc(BSH * 2);
  unsigned short* wp[6];
  for (int i = 0; i < 6; ++i) wp[i] = (unsigned short*)alloc((size_t)16 * 24 * 512 * 2);
  float*          xq   = (float*)alloc(BSH * 4);
  unsigned short* ping = (unsigned short*)alloc(BS * 256 * 2);
  unsigned short* pong = (unsigned short*)alloc(BS * 256 * 2);
  int*   eidx  = (int*)alloc(BS * 4);
  int*   cum   = (int*)alloc(BS * 4);
  int*   mlen  = (int*)alloc(BB * 4);
  int*   fidx  = (int*)alloc(BT * 4);
  int*   vmask = (int*)alloc(BT * 4);
  unsigned short* xexp = (unsigned short*)alloc(BTH * 2);
  unsigned short* h1p  = (unsigned short*)alloc(BTH * 2);
  unsigned short* h2p  = (unsigned short*)alloc(BTH * 2);
  float* p1d = (float*)alloc(BT * 4);
  float* sbh = (float*)alloc((size_t)BB * 256 * 4);

  // d_out layout (tuple order, all float)
  float* out = (float*)d_out;
  const size_t o_xout  = 0;
  const size_t o_pitch = o_xout + BTH;
  const size_t o_mv    = o_pitch + BT * NSN;
  const size_t o_en    = o_mv + (size_t)BB * 2;
  const size_t o_ld    = o_en + BS;
  const size_t o_dur   = o_ld + BS;
  const size_t o_mlen  = o_dur + BS;
  const size_t o_mmask = o_mlen + BB;

  k_cast<<<(unsigned)((BSH + 255) / 256), 256, 0, stream>>>(x, xbf, (int)BSH);
  for (int i = 0; i < 6; ++i)
    k_pack<<<768, 256, 0, stream>>>(conv_w[i], wp[i]);

  // duration predictor
  k_conv_ln<<<(unsigned)(BS / 16), 256, 0, stream>>>(xbf,  wp[0], conv_b[0], conv_g[0], conv_be[0], ping, SS);
  k_conv_ln<<<(unsigned)(BS / 16), 256, 0, stream>>>(ping, wp[1], conv_b[1], conv_g[1], conv_be[1], pong, SS);
  k_rowdot<<<(unsigned)(BS / 8), 256, 0, stream>>>(pong, dp_lw, dp_lb, src_mask, out + o_ld, (int)BS);
  // energy predictor
  k_conv_ln<<<(unsigned)(BS / 16), 256, 0, stream>>>(xbf,  wp[2], conv_b[2], conv_g[2], conv_be[2], ping, SS);
  k_conv_ln<<<(unsigned)(BS / 16), 256, 0, stream>>>(ping, wp[3], conv_b[3], conv_g[3], conv_be[3], pong, SS);
  k_rowdot<<<(unsigned)(BS / 8), 256, 0, stream>>>(pong, ep_lw, ep_lb, src_mask, out + o_en, (int)BS);

  // energy embedding + length regulate
  k_eidx<<<(unsigned)(BS / 256), 256, 0, stream>>>(etgt, bins, eidx, (int)BS);
  k_cum<<<1, BB, 0, stream>>>(dur, cum, mlen, out + o_mlen);
  k_addemb<<<(unsigned)(BSH / 256), 256, 0, stream>>>(x, emb, eidx, xq, (int)BSH);
  k_expand_idx<<<(unsigned)(BT / 256), 256, 0, stream>>>(cum, mlen, fidx, vmask, (int)BT);
  k_gather<<<(unsigned)(BTH / 256), 256, 0, stream>>>(xq, fidx, vmask, xexp, (int)BTH);

  // postnet convs (the dominant 51.5 GFLOP -> WMMA)
  k_conv_ln<<<(unsigned)(BT / 16), 256, 0, stream>>>(xexp, wp[4], conv_b[4], conv_g[4], conv_be[4], h1p, TT);
  k_conv_ln<<<(unsigned)(BT / 16), 256, 0, stream>>>(h1p,  wp[5], conv_b[5], conv_g[5], conv_be[5], h2p, TT);

  // heads
  k_pitch<<<(unsigned)(BT / 8), 256, 0, stream>>>(h2p, pp_w, pp_b, icwt, out + o_pitch, p1d, (int)BT);
  k_hsum<<<BB, 256, 0, stream>>>(h2p, sbh);
  k_meanvar<<<1, 32, 0, stream>>>(sbh, ps_w, ps_b, out + o_mv);
  k_xout<<<(unsigned)(BTH / 256), 256, 0, stream>>>(xq, fidx, vmask, p1d, out + o_xout, (int)BTH);
  k_misc<<<(unsigned)(BT / 256), 256, 0, stream>>>(dur, mel_mask, out + o_dur, out + o_mmask);
}
